// MultiHeadAttention2_43026982371479
// MI455X (gfx1250) — compile-verified, matrix operand order checked by reading, not run on hardware
//
#include <hip/hip_runtime.h>

typedef __bf16 bf16;
typedef __attribute__((ext_vector_type(16))) __bf16 v16bf;
typedef __attribute__((ext_vector_type(8)))  __bf16 v8bf;
typedef __attribute__((ext_vector_type(8)))  float  v8f;

#define N_TOK  4096
#define DMODEL 1024
#define NHEAD  16
#define DH     64

// ---------------------------------------------------------------------------
// WMMA helper: D = A(16x32 bf16) * B(32x16 bf16) + C(16x16 f32)
// ---------------------------------------------------------------------------
__device__ __forceinline__ v8f wmma_bf16(v16bf a, v16bf b, v8f c) {
  return __builtin_amdgcn_wmma_f32_16x16x32_bf16(
      /*neg_a=*/false, a, /*neg_b=*/false, b,
      /*c_mod=*/(short)0, c, /*reuse_a=*/false, /*reuse_b=*/false);
}

// Load one 16-lane-striped bf16 fragment row-slice.
// ISA 16-bit A layout (B mirrors with lane=column):
//   lanes 0-15 : K = k0+0..7 (elems 0..7),  k0+16..23 (elems 8..15)
//   lanes 16-31: K = k0+8..15,             k0+24..31
// -> two contiguous 16B loads per lane.
__device__ __forceinline__ v16bf load_frag(const bf16* __restrict__ row,
                                           int k0, int lane) {
  int kb = k0 + ((lane & 16) ? 8 : 0);
  v8bf lo = *(const v8bf*)(row + kb);
  v8bf hi = *(const v8bf*)(row + kb + 16);
  v16bf r;
#pragma unroll
  for (int i = 0; i < 8; ++i) { r[i] = lo[i]; r[i + 8] = hi[i]; }
  return r;
}

__device__ __forceinline__ v8bf cvt8(v8f a) {
  v8bf r;
#pragma unroll
  for (int i = 0; i < 8; ++i) r[i] = (bf16)a[i];
  return r;
}

// ---------------------------------------------------------------------------
// f32 -> bf16 conversion, 8 elements / thread
// ---------------------------------------------------------------------------
__global__ __launch_bounds__(256) void cvt_f32_bf16(const float* __restrict__ in,
                                                    bf16* __restrict__ out,
                                                    int n8) {
  int i = blockIdx.x * blockDim.x + threadIdx.x;
  if (i >= n8) return;
  int base = i * 8;
  float4 x = *(const float4*)(in + base);
  float4 y = *(const float4*)(in + base + 4);
  v8bf o;
  o[0] = (bf16)x.x; o[1] = (bf16)x.y; o[2] = (bf16)x.z; o[3] = (bf16)x.w;
  o[4] = (bf16)y.x; o[5] = (bf16)y.y; o[6] = (bf16)y.z; o[7] = (bf16)y.w;
  *(v8bf*)(out + base) = o;
}

// ---------------------------------------------------------------------------
// Projection GEMM. Y = X @ W^T + b, one 64x64 tile per wave, 8 waves/block.
// MODE 0 (Q,K): tile M-dim = d (A = W rows), N-dim = n (B = X rows).
//               out[h][n][dh] bf16  (h = d>>6) -> contiguous b128 stores.
// MODE 1 (V):   tile M-dim = n (A = X rows), N-dim = d (B = W rows).
//               out[h][dh][n] bf16  (V pre-transposed) -> contiguous stores.
// ---------------------------------------------------------------------------
template <int MODE>
__global__ __launch_bounds__(256) void proj_kernel(const bf16* __restrict__ Amat,
                                                   const bf16* __restrict__ Bmat,
                                                   const float* __restrict__ bias,
                                                   bf16* __restrict__ out) {
  const int lane = threadIdx.x & 31;
  const int wave = threadIdx.x >> 5;
  const int lm   = lane & 15;
  const int moff = (lane & 16) ? 8 : 0;

  const int tiles_m = (MODE == 0) ? (DMODEL / 64) : (N_TOK / 64);
  const int id = blockIdx.x * 8 + wave;
  const int tm = id % tiles_m;
  const int tn = id / tiles_m;

  v8f acc[4][4] = {};

#pragma unroll 2
  for (int k0 = 0; k0 < DMODEL; k0 += 32) {
    v16bf a[4], b[4];
#pragma unroll
    for (int mt = 0; mt < 4; ++mt)
      a[mt] = load_frag(Amat + (size_t)(tm * 64 + mt * 16 + lm) * DMODEL, k0, lane);
#pragma unroll
    for (int nt = 0; nt < 4; ++nt)
      b[nt] = load_frag(Bmat + (size_t)(tn * 64 + nt * 16 + lm) * DMODEL, k0, lane);
#pragma unroll
    for (int mt = 0; mt < 4; ++mt)
#pragma unroll
      for (int nt = 0; nt < 4; ++nt)
        acc[mt][nt] = wmma_bf16(a[mt], b[nt], acc[mt][nt]);
  }

  if (MODE == 0) {
#pragma unroll
    for (int mt = 0; mt < 4; ++mt)
#pragma unroll
      for (int r = 0; r < 8; ++r) {
        float bb = bias[tm * 64 + mt * 16 + moff + r];
#pragma unroll
        for (int nt = 0; nt < 4; ++nt) acc[mt][nt][r] += bb;
      }
#pragma unroll
    for (int mt = 0; mt < 4; ++mt)
#pragma unroll
      for (int nt = 0; nt < 4; ++nt) {
        int n = tn * 64 + nt * 16 + lm;
        size_t off = ((size_t)(tm * N_TOK + n)) * DH + mt * 16 + moff;
        *(v8bf*)(out + off) = cvt8(acc[mt][nt]);
      }
  } else {
#pragma unroll
    for (int nt = 0; nt < 4; ++nt) {
      float bb = bias[tn * 64 + nt * 16 + lm];
#pragma unroll
      for (int mt = 0; mt < 4; ++mt)
#pragma unroll
        for (int r = 0; r < 8; ++r) acc[mt][nt][r] += bb;
    }
#pragma unroll
    for (int mt = 0; mt < 4; ++mt)
#pragma unroll
      for (int nt = 0; nt < 4; ++nt) {
        int d = tn * 64 + nt * 16 + lm;
        size_t off = (size_t)d * N_TOK + tm * 64 + mt * 16 + moff;
        *(v8bf*)(out + off) = cvt8(acc[mt][nt]);
      }
  }
}

// ---------------------------------------------------------------------------
// Flash attention. Block = (head, 128 Q rows), 4 waves x 32 rows (2 M-tiles).
// Every K/V fragment is reused by both M-tiles -> 2x arithmetic intensity.
// Row softmax stats per half-wave: element r <-> row r + 8*(lane>=16).
// Row sums computed by WMMA against an all-ones B fragment (no shuffles).
// ---------------------------------------------------------------------------
__device__ __forceinline__ float half_max(float v) {
#pragma unroll
  for (int m = 1; m < 16; m <<= 1) v = fmaxf(v, __shfl_xor(v, m, 32));
  return v;
}

#define PPITCH 72  // padded LDS row pitch (bf16 elems) to spread banks

__global__ __launch_bounds__(128) void attn_kernel(const bf16* __restrict__ Qb,
                                                   const bf16* __restrict__ Kb,
                                                   const bf16* __restrict__ VTb,
                                                   float* __restrict__ out) {
  __shared__ bf16 pstage[4 * 32 * PPITCH];

  const int h    = blockIdx.y;
  const int qblk = blockIdx.x;
  const int lane = threadIdx.x & 31;
  const int wave = threadIdx.x >> 5;
  const int lm   = lane & 15;
  const int moff = (lane & 16) ? 8 : 0;

  const bf16* Qh = Qb  + (size_t)h * N_TOK * DH;
  const bf16* Kh = Kb  + (size_t)h * N_TOK * DH;
  const bf16* Vh = VTb + (size_t)h * DH * N_TOK;  // transposed [dh][n]

  // Q fragments, pre-scaled by 1/sqrt(Dh) = 2^-3 (exact in bf16)
  const int qrow0 = qblk * 128 + wave * 32;
  v16bf qa[2][2];
#pragma unroll
  for (int mt = 0; mt < 2; ++mt)
#pragma unroll
    for (int s = 0; s < 2; ++s) {
      v16bf q = load_frag(Qh + (size_t)(qrow0 + mt * 16 + lm) * DH, s * 32, lane);
#pragma unroll
      for (int i = 0; i < 16; ++i) q[i] = (bf16)(0.125f * (float)q[i]);
      qa[mt][s] = q;
    }

  // all-ones B fragment for row-sum WMMA
  v16bf ones;
#pragma unroll
  for (int i = 0; i < 16; ++i) ones[i] = (bf16)1.0f;

  float m_[2][8], l_[2][8];
  v8f oacc[2][4] = {};
#pragma unroll
  for (int mt = 0; mt < 2; ++mt)
#pragma unroll
    for (int r = 0; r < 8; ++r) { m_[mt][r] = -3.0e38f; l_[mt][r] = 0.0f; }

  const int wbase = wave * 32 * PPITCH;

#pragma unroll 1
  for (int jt = 0; jt < N_TOK / 64; ++jt) {
    // ---- S = (Q/8) K^T : each K fragment feeds both M-tiles ----
    v8f sacc[2][4] = {};
#pragma unroll
    for (int t = 0; t < 4; ++t) {
      const bf16* kp = Kh + (size_t)(jt * 64 + t * 16 + lm) * DH;
      v16bf b0 = load_frag(kp, 0, lane);
      v16bf b1 = load_frag(kp, 32, lane);
#pragma unroll
      for (int mt = 0; mt < 2; ++mt) {
        sacc[mt][t] = wmma_bf16(qa[mt][0], b0, sacc[mt][t]);
        sacc[mt][t] = wmma_bf16(qa[mt][1], b1, sacc[mt][t]);
      }
    }

    // ---- online softmax: row max (shuffle) + exp ----
    float alpha[2][8];
#pragma unroll
    for (int mt = 0; mt < 2; ++mt)
#pragma unroll
      for (int r = 0; r < 8; ++r) {
        float v = fmaxf(fmaxf(sacc[mt][0][r], sacc[mt][1][r]),
                        fmaxf(sacc[mt][2][r], sacc[mt][3][r]));
        v = half_max(v);
        float mn = fmaxf(m_[mt][r], v);
        alpha[mt][r] = __expf(m_[mt][r] - mn);
        m_[mt][r] = mn;
      }
#pragma unroll
    for (int mt = 0; mt < 2; ++mt)
#pragma unroll
      for (int t = 0; t < 4; ++t)
#pragma unroll
        for (int r = 0; r < 8; ++r)
          sacc[mt][t][r] = __expf(sacc[mt][t][r] - m_[mt][r]);

    // ---- relayout P: accum layout -> A-fragment layout via LDS ----
    __syncthreads();
#pragma unroll
    for (int mt = 0; mt < 2; ++mt)
#pragma unroll
      for (int t = 0; t < 4; ++t)
#pragma unroll
        for (int r = 0; r < 8; ++r)
          pstage[wbase + (mt * 16 + r + moff) * PPITCH + t * 16 + lm] =
              (bf16)sacc[mt][t][r];
    __syncthreads();

    v16bf pa[2][2];
#pragma unroll
    for (int mt = 0; mt < 2; ++mt) {
      const bf16* prow = &pstage[wbase + (mt * 16 + lm) * PPITCH];
      pa[mt][0] = load_frag(prow, 0, lane);
      pa[mt][1] = load_frag(prow, 32, lane);
    }

    // ---- row sums via WMMA with ones-matrix (accum layout matches stats) ----
#pragma unroll
    for (int mt = 0; mt < 2; ++mt) {
      v8f lsum = {};
      lsum = wmma_bf16(pa[mt][0], ones, lsum);
      lsum = wmma_bf16(pa[mt][1], ones, lsum);
#pragma unroll
      for (int r = 0; r < 8; ++r) {
        l_[mt][r] = l_[mt][r] * alpha[mt][r] + lsum[r];
#pragma unroll
        for (int t = 0; t < 4; ++t) oacc[mt][t][r] *= alpha[mt][r];
      }
    }

    // ---- O += P V : each V fragment feeds both M-tiles ----
#pragma unroll
    for (int t = 0; t < 4; ++t) {
      const bf16* vp = Vh + (size_t)(t * 16 + lm) * N_TOK + jt * 64;
      v16bf vb0 = load_frag(vp, 0, lane);
      v16bf vb1 = load_frag(vp, 32, lane);
#pragma unroll
      for (int mt = 0; mt < 2; ++mt) {
        oacc[mt][t] = wmma_bf16(pa[mt][0], vb0, oacc[mt][t]);
        oacc[mt][t] = wmma_bf16(pa[mt][1], vb1, oacc[mt][t]);
      }
    }
  }

  // ---- finalize: O / l, write f32 [N][1][D] ----
#pragma unroll
  for (int mt = 0; mt < 2; ++mt) {
    float inv[8];
#pragma unroll
    for (int r = 0; r < 8; ++r) inv[r] = 1.0f / l_[mt][r];
#pragma unroll
    for (int t = 0; t < 4; ++t)
#pragma unroll
      for (int r = 0; r < 8; ++r) {
        int row = qrow0 + mt * 16 + moff + r;
        out[(size_t)row * DMODEL + h * DH + t * 16 + lm] = oacc[mt][t][r] * inv[r];
      }
  }
}

// ---------------------------------------------------------------------------
// Launch
// ---------------------------------------------------------------------------
extern "C" void kernel_launch(void* const* d_in, const int* in_sizes, int n_in,
                              void* d_out, int out_size, void* d_ws, size_t ws_size,
                              hipStream_t stream) {
  const float* t1 = (const float*)d_in[0];
  const float* t2 = (const float*)d_in[1];
  const float* Wq = (const float*)d_in[2];
  const float* bq = (const float*)d_in[3];
  const float* Wk = (const float*)d_in[4];
  const float* bk = (const float*)d_in[5];
  const float* Wv = (const float*)d_in[6];
  const float* bv = (const float*)d_in[7];
  float* out = (float*)d_out;

  bf16* ws  = (bf16*)d_ws;
  bf16* Xt2 = ws;                    // [4096][1024] bf16
  bf16* Xt1 = Xt2 + 4194304;
  bf16* Wqb = Xt1 + 4194304;         // [1024][1024] bf16 each
  bf16* Wkb = Wqb + 1048576;
  bf16* Wvb = Wkb + 1048576;
  bf16* Qb  = Wvb + 1048576;         // [H][N][64]
  bf16* Kb  = Qb  + 4194304;         // [H][N][64]
  bf16* VTb = Kb  + 4194304;         // [H][64][N]  (transposed V)

  cvt_f32_bf16<<<2048, 256, 0, stream>>>(t2, Xt2, 524288);
  cvt_f32_bf16<<<2048, 256, 0, stream>>>(t1, Xt1, 524288);
  cvt_f32_bf16<<<512,  256, 0, stream>>>(Wq, Wqb, 131072);
  cvt_f32_bf16<<<512,  256, 0, stream>>>(Wk, Wkb, 131072);
  cvt_f32_bf16<<<512,  256, 0, stream>>>(Wv, Wvb, 131072);

  proj_kernel<0><<<128, 256, 0, stream>>>(Wqb, Xt2, bq, Qb);   // Q = t2 Wq^T + bq
  proj_kernel<0><<<128, 256, 0, stream>>>(Wkb, Xt1, bk, Kb);   // K = t1 Wk^T + bk
  proj_kernel<1><<<128, 256, 0, stream>>>(Xt1, Wvb, bv, VTb);  // V^T

  attn_kernel<<<dim3(N_TOK / 128, NHEAD), 128, 0, stream>>>(Qb, Kb, VTb, out);
}
